// DensityAwareFeatureAggregator_42417097015678
// MI455X (gfx1250) — matrix operand order repaired and checked by reading, not compiled, not run on hardware
//
#include <hip/hip_runtime.h>

typedef __attribute__((ext_vector_type(16))) __bf16 bf16x16;
typedef __attribute__((ext_vector_type(8)))  __bf16 bf16x8;
typedef __attribute__((ext_vector_type(8)))  float  f32x8;

union Frag { bf16x16 v; bf16x8 h[2]; };

static constexpr int NPTS = 16384;   // N
static constexpr int PSTR = 72;      // LDS row stride (bf16) for 32x64 staging (+8 pad)
static constexpr int VSTR = 136;     // LDS row stride (bf16) for 16x128 mean-vector staging (+8 pad)

__device__ __forceinline__ f32x8 wmma_bf16(const Frag& a, const Frag& b, f32x8 c) {
  return __builtin_amdgcn_wmma_f32_16x16x32_bf16(false, a.v, false, b.v,
                                                 (short)0, c, false, false);
}

// ---------------- prep kernels (rerun every launch; d_ws is not persistent) ----------------

__global__ void k_pack_feat(const float* __restrict__ f, __bf16* __restrict__ o, int n) {
  int i = blockIdx.x * 256 + threadIdx.x;
  if (i < n) o[i] = (__bf16)f[i];
}

// Pack W (K x ncols, row-major f32) into B-fragment-linear bf16:
// dst[(t*32+L)*16+e] = bf16( W[(kc*32 + (e>>3)*16 + (L>>4)*8 + (e&7))*ncols + nt*16 + (L&15)] ), t = nt*KC+kc
__global__ void k_pack_w(const float* __restrict__ w, __bf16* __restrict__ dst,
                         int KC, int NT, int ncols) {
  int i = blockIdx.x * 256 + threadIdx.x;
  if (i >= NT * KC * 512) return;
  int e  = i & 15;
  int L  = (i >> 4) & 31;
  int t  = i >> 9;
  int kc = t % KC;
  int nt = t / KC;
  int k   = kc * 32 + ((e >> 3) << 4) + ((L >> 4) << 3) + (e & 7);
  int col = nt * 16 + (L & 15);
  dst[i] = (__bf16)w[k * ncols + col];
}

// pe2 has no relu after it -> its bias is purely affine and folds through mlp1:
// b1_eff[j] = mb1[j] + sum_i peb2[i] * mw1[(32+i)*128 + j]
__global__ void k_bias(const float* __restrict__ mb1, const float* __restrict__ peb2,
                       const float* __restrict__ mw1, float* __restrict__ b1eff) {
  int j = threadIdx.x;   // 128 threads
  float s = mb1[j];
  for (int i = 0; i < 64; ++i) s = fmaf(peb2[i], mw1[(32 + i) * 128 + j], s);
  b1eff[j] = s;
}

// ---------------- main kernel: one wave == one strip of 16 points ----------------
// launch_bounds(128, 4): cap allocation at <=256 VGPRs -> no s_set_vgpr_msb, >=4 waves/SIMD

__global__ __launch_bounds__(128, 4) void k_main(
    const float* __restrict__ points, const int* __restrict__ nidx,
    const float* __restrict__ pew1, const float* __restrict__ peb1,
    const float* __restrict__ b1eff, const float* __restrict__ mb2,
    const __bf16* __restrict__ featBF,
    const __bf16* __restrict__ pPE2,
    const __bf16* __restrict__ pM1,
    const __bf16* __restrict__ pM2,
    float* __restrict__ out) {
  __shared__ alignas(16) __bf16 ldsP[4][32 * PSTR];  // 18.0 KB : pe1-out / pe2-out staging
  __shared__ alignas(16) __bf16 ldsV[4][16 * VSTR];  // 17.0 KB : 16 mean vectors (16x128)

  const int lane = threadIdx.x & 31;
  const int wave = threadIdx.x >> 5;
  const int colL = lane & 15;
  const int hi   = lane >> 4;          // 0 or 1
  const int p0   = (blockIdx.x * 4 + wave) * 16;   // strip base point id (= b*N+n)

  __bf16* bufP = ldsP[wave];
  __bf16* bufV = ldsV[wave];

  // per-lane base pointers -> all ds/global ops below use constant immediate offsets
  const __bf16* aP = bufP + colL * PSTR + (hi << 3);         // A-frag loads from bufP
  __bf16*       dP = bufP + (hi << 3) * PSTR + colL;         // D-tile scatter into bufP
  const __bf16* vA = bufV + colL * VSTR + (hi << 3);         // mlp2 A-frag loads
  __bf16*       vS = bufV + colL;                            // mean-vector stores
  const int laneOff = lane * 2;                              // bf16x8 units into packed B

  for (int pt = 0; pt < 16; ++pt) {
    const int p = p0 + pt;
    const int b = p >> 14;

    // Opaque zero SGPR offset, renewed per iteration and added to the UNIFORM base:
    // blocks LICM of the loop-invariant B-fragment loads (prevents 256-VGPR hoist +
    // scratch spill) while keeping a uniform SGPR base + divergent 32-bit offset so
    // the backend selects global_load with saddr form.
    int licm = 0;
    asm volatile("" : "+s"(licm));
    const bf16x8* bP2 = ((const bf16x8*)pPE2 + licm) + laneOff;
    const bf16x8* bM1 = ((const bf16x8*)pM1  + licm) + laneOff;

    // ---- gather rel_pos + pe1 = relu(rel @ W1(3x64) + b1) on VALU; lane = neighbor k ----
    const int    idxk = nidx[p * 32 + lane];
    const float* pc   = points + p * 3;
    const float* pk   = points + ((b << 14) + idxk) * 3;
    const float  rx = pk[0] - pc[0], ry = pk[1] - pc[1], rz = pk[2] - pc[2];

#pragma unroll
    for (int s = 0; s < 8; ++s) {
      bf16x8 q;
#pragma unroll
      for (int d = 0; d < 8; ++d) {
        const int j = s * 8 + d;
        float a = fmaf(rz, pew1[128 + j], fmaf(ry, pew1[64 + j], fmaf(rx, pew1[j], peb1[j])));
        q[d] = (__bf16)fmaxf(a, 0.f);
      }
      *(bf16x8*)(bufP + lane * PSTR + s * 8) = q;   // row k of 32x64 bf16
    }

    // ---- pe2: (32x64) @ W2(64x64), bias folded into b1eff; zero C -> inline 0 SRC2 ----
    Frag Ap[2][2];
#pragma unroll
    for (int m = 0; m < 2; ++m)
#pragma unroll
      for (int kc = 0; kc < 2; ++kc) {
        Ap[m][kc].h[0] = *(const bf16x8*)(aP + m * 16 * PSTR + kc * 32);
        Ap[m][kc].h[1] = *(const bf16x8*)(aP + m * 16 * PSTR + kc * 32 + 16);
      }

#pragma unroll
    for (int nt = 0; nt < 4; ++nt) {
      f32x8 a0 = {}, a1 = {};
#pragma unroll
      for (int kc = 0; kc < 2; ++kc) {
        const int t = nt * 2 + kc;
        Frag B;
        B.h[0] = bP2[t * 64];
        B.h[1] = bP2[t * 64 + 1];
        a0 = wmma_bf16(Ap[0][kc], B, a0);
        a1 = wmma_bf16(Ap[1][kc], B, a1);
      }
#pragma unroll
      for (int r = 0; r < 8; ++r) {                 // C/D layout -> row-major bufP
        dP[r * PSTR + nt * 16]        = (__bf16)a0[r];
        dP[(16 + r) * PSTR + nt * 16] = (__bf16)a1[r];
      }
    }

    // ---- mlp1: cat(32x96) @ W(96x128) + b1eff, relu, then mean over the 32 rows ----
    Frag Ac[2][3];
#pragma unroll
    for (int m = 0; m < 2; ++m) {
      const int ridx = nidx[p * 32 + colL + 16 * m];
      const __bf16* fb = featBF + ((b << 14) + ridx) * 32 + (hi << 3);
      Ac[m][0].h[0] = *(const bf16x8*)fb;
      Ac[m][0].h[1] = *(const bf16x8*)(fb + 16);
      Ac[m][1].h[0] = *(const bf16x8*)(aP + m * 16 * PSTR);
      Ac[m][1].h[1] = *(const bf16x8*)(aP + m * 16 * PSTR + 16);
      Ac[m][2].h[0] = *(const bf16x8*)(aP + m * 16 * PSTR + 32);
      Ac[m][2].h[1] = *(const bf16x8*)(aP + m * 16 * PSTR + 48);
    }

#pragma unroll
    for (int nt = 0; nt < 8; ++nt) {
      const float bias = b1eff[nt * 16 + colL];
      f32x8 a0, a1;
#pragma unroll
      for (int r = 0; r < 8; ++r) { a0[r] = bias; a1[r] = bias; }
#pragma unroll
      for (int kc = 0; kc < 3; ++kc) {
        const int t = nt * 3 + kc;
        Frag B;
        B.h[0] = bM1[t * 64];
        B.h[1] = bM1[t * 64 + 1];
        a0 = wmma_bf16(Ac[0][kc], B, a0);
        a1 = wmma_bf16(Ac[1][kc], B, a1);
      }
      // relu + pairwise-tree row reduction (mean commutes with the affine mlp2)
      float t0[8];
#pragma unroll
      for (int r = 0; r < 8; ++r) t0[r] = fmaxf(a0[r], 0.f) + fmaxf(a1[r], 0.f);
#pragma unroll
      for (int r = 0; r < 4; ++r) t0[r] += t0[r + 4];
      t0[0] += t0[2];
      t0[1] += t0[3];
      float s = t0[0] + t0[1];
      s += __shfl_xor(s, 16, 32);   // both half-lanes now hold the identical 32-row sum
      vS[pt * VSTR + nt * 16] = (__bf16)(s * 0.03125f);  // duplicate same-value store is benign
    }
  }

  // ---- mlp2 for the whole strip: 16 mean-vectors (16x128) @ W(128x64) + b ----
  const bf16x8* bM2 = (const bf16x8*)pM2 + laneOff;
  Frag Av[4];
#pragma unroll
  for (int kc = 0; kc < 4; ++kc) {
    Av[kc].h[0] = *(const bf16x8*)(vA + kc * 32);
    Av[kc].h[1] = *(const bf16x8*)(vA + kc * 32 + 16);
  }

  float* op = out + (p0 + (hi << 3)) * 64 + colL;   // row = point-in-strip, col = feature
#pragma unroll
  for (int nt = 0; nt < 4; ++nt) {
    const float bias = mb2[nt * 16 + colL];
    f32x8 acc;
#pragma unroll
    for (int r = 0; r < 8; ++r) acc[r] = bias;
#pragma unroll
    for (int kc = 0; kc < 4; ++kc) {
      const int t = nt * 4 + kc;
      Frag B;
      B.h[0] = bM2[t * 64];
      B.h[1] = bM2[t * 64 + 1];
      acc = wmma_bf16(Av[kc], B, acc);
    }
#pragma unroll
    for (int r = 0; r < 8; ++r) op[r * 64 + nt * 16] = acc[r];
  }
}

extern "C" void kernel_launch(void* const* d_in, const int* in_sizes, int n_in,
                              void* d_out, int out_size, void* d_ws, size_t ws_size,
                              hipStream_t stream) {
  const float* points   = (const float*)d_in[0];
  const float* features = (const float*)d_in[1];
  // d_in[2] = density : softmax over K of a per-point constant == 1/K -> density MLP cancels
  const int*   nidx     = (const int*)d_in[3];
  const float* pew1     = (const float*)d_in[4];
  const float* peb1     = (const float*)d_in[5];
  const float* pew2     = (const float*)d_in[6];
  const float* peb2     = (const float*)d_in[7];
  const float* mw1      = (const float*)d_in[8];
  const float* mb1      = (const float*)d_in[9];
  const float* mw2      = (const float*)d_in[10];
  const float* mb2      = (const float*)d_in[11];
  float*       out      = (float*)d_out;

  // workspace layout (bytes)
  __bf16* featBF = (__bf16*)d_ws;                                    // 4 MiB
  __bf16* pPE2   = (__bf16*)((char*)d_ws + 4194304);                 // 8 KiB
  __bf16* pM1    = (__bf16*)((char*)d_ws + 4194304 + 8192);          // 24 KiB
  __bf16* pM2    = (__bf16*)((char*)d_ws + 4194304 + 8192 + 24576);  // 16 KiB
  float*  b1eff  = (float*)((char*)d_ws + 4194304 + 8192 + 24576 + 16384); // 512 B

  const int nfeat = 4 * 16384 * 32;
  k_pack_feat<<<(nfeat + 255) / 256, 256, 0, stream>>>(features, featBF, nfeat);
  k_pack_w<<<(4 * 2 * 512 + 255) / 256, 256, 0, stream>>>(pew2, pPE2, 2, 4, 64);
  k_pack_w<<<(8 * 3 * 512 + 255) / 256, 256, 0, stream>>>(mw1,  pM1,  3, 8, 128);
  k_pack_w<<<(4 * 4 * 512 + 255) / 256, 256, 0, stream>>>(mw2,  pM2,  4, 4, 64);
  k_bias<<<1, 128, 0, stream>>>(mb1, peb2, mw1, b1eff);

  // 65536 points, 4 waves/block, 16 points/wave -> 1024 blocks
  k_main<<<1024, 128, 0, stream>>>(points, nidx, pew1, peb1, b1eff, mb2,
                                   featBF, pPE2, pM1, pM2, out);
}
